// SO3Convolution_27513560498716
// MI455X (gfx1250) — matrix operand order repaired and checked by reading, not compiled, not run on hardware
//
#include <hip/hip_runtime.h>
#include <hip/hip_bf16.h>
#include <math.h>

// ---------------------------------------------------------------------------
// SO3Convolution for MI455X (gfx1250, wave32, WMMA f16->f32, async-to-LDS)
//
// Phase 1: psi^T per-l  = (w[16384x512] @ D[512x455]) / sqrt(512), stored f16
//          in d_ws as per-l transposed weight matrices W_l^T[(g,v)][(f,u)].
//          Needs 16384*455*2 = 14.9 MB of workspace.
// Phase 2: per l, per i: Y_i[256 x 128d] = X_i[256 x 128d] @ W_l[128d x 128d]
//          scaled by 1/sqrt(128d), written to out[b,g, off + v*d + i].
//          B tiles streamed with GLOBAL_LOAD_ASYNC_TO_LDS_B128 (ASYNCcnt).
// ---------------------------------------------------------------------------

typedef __attribute__((ext_vector_type(16))) _Float16 v16h;
typedef __attribute__((ext_vector_type(8)))  float    v8f;

#define IRREPS 455
#define NGRID  512

__device__ __constant__ int c_OFF[8] = {0, 1, 10, 35, 84, 165, 286, 455};

// low 32 bits of a generic LDS pointer == wave-relative LDS byte offset
__device__ __forceinline__ unsigned lds_off32(const void* p) {
  return (unsigned)(uintptr_t)p;
}

// ============================ Phase 1 ======================================
// grid: 128 blocks (128 M-rows each), block: 256 threads = 8 waves.
// dynamic LDS: wA[128][520] f16 + dT[16][520] f16  (~150 KB, WGP has 320 KB)
#define P1_PITCH 520   // 512 + 8 halves padding (breaks bank aliasing)

__global__ __launch_bounds__(256) void psi_kernel(
    const float* __restrict__ w,    // [16384, 512]
    const float* __restrict__ Dm,   // [512, 455]
    _Float16*    __restrict__ psiT) // workspace, per-l W^T, f16
{
  extern __shared__ _Float16 smem[];
  _Float16* wA = smem;                      // [128][P1_PITCH]
  _Float16* dT = smem + 128 * P1_PITCH;     // [16][P1_PITCH]

  const int tid  = threadIdx.x;
  const int m0   = blockIdx.x * 128;
  const int wave = tid >> 5;
  const int lane = tid & 31;
  const int lh   = lane >> 4;   // 0/1 : K-half select
  const int ll   = lane & 15;   // row (A) / col (B,D)

  // --- stage 128x512 strip of w into LDS as f16 (read w from HBM once) ---
  {
    const float4* wsrc = (const float4*)(w + (size_t)m0 * NGRID);
    #pragma unroll
    for (int j = 0; j < 64; ++j) {
      int e  = tid + 256 * j;     // 0..16383 float4s
      int r  = e >> 7;            // row 0..127  (128 float4 per row)
      int c4 = e & 127;
      float4 v = wsrc[(size_t)r * 128 + c4];
      _Float16* dst = wA + r * P1_PITCH + c4 * 4;
      dst[0] = (_Float16)v.x; dst[1] = (_Float16)v.y;
      dst[2] = (_Float16)v.z; dst[3] = (_Float16)v.w;
    }
  }

  const float scale1 = 0.044194173824159216f; // 1/sqrt(512)
  const int arow = wave * 16 + ll;            // this lane's A row in strip

  for (int nt = 0; nt < 29; ++nt) {           // ceil(455/16) N-tiles
    __syncthreads();                          // protect dT from prev iter
    const int i0 = nt * 16;
    // --- load D[:, i0:i0+16] transposed into dT[16][512] (f16) ---
    #pragma unroll
    for (int j = 0; j < 32; ++j) {
      int e = tid + 256 * j;      // 0..8191
      int c = e & 15;             // col within tile
      int k = e >> 4;             // 0..511
      float v = 0.0f;
      if (i0 + c < IRREPS) v = Dm[(size_t)k * IRREPS + i0 + c];
      dT[c * P1_PITCH + k] = (_Float16)v;
    }
    __syncthreads();

    // fragment double-buffer so LDS loads for kc+1 overlap wmma kc
    auto ldfrag = [&](v16h& a, v16h& b, int kc) {
      const _Float16* ap = wA + arow * P1_PITCH + kc * 32 + lh * 8;
      ((uint4*)&a)[0] = *(const uint4*)(ap);        // K 0..7   (or 8..15)
      ((uint4*)&a)[1] = *(const uint4*)(ap + 16);   // K 16..23 (or 24..31)
      const _Float16* bp = dT + ll * P1_PITCH + kc * 32 + lh * 8;
      ((uint4*)&b)[0] = *(const uint4*)(bp);        // K 0..7   (or 8..15)
      ((uint4*)&b)[1] = *(const uint4*)(bp + 16);   // K 16..23 (or 24..31)
    };

    v8f acc = {};
    v16h a[2], b[2];
    ldfrag(a[0], b[0], 0);
    #pragma unroll
    for (int kc = 0; kc < 16; ++kc) {         // K = 512 = 16 x 32
      const int cur = kc & 1;
      if (kc + 1 < 16) ldfrag(a[cur ^ 1], b[cur ^ 1], kc + 1);
      acc = __builtin_amdgcn_wmma_f32_16x16x32_f16(
          false, a[cur], false, b[cur], (short)0, acc, false, false);
    }

    // --- scatter-store as per-l transposed weights, f16, scaled ---
    const int i = i0 + ll;                    // this lane's psi column
    if (i < IRREPS) {
      int l = 0;
      while (i >= c_OFF[l + 1]) ++l;
      const int d  = 2 * l + 1;
      const int ii = i - c_OFF[l];
      const int u  = ii / d;
      const int v  = ii - u * d;
      const size_t base = (size_t)c_OFF[l] * 16384;
      const int Kl = 128 * d;
      #pragma unroll
      for (int r = 0; r < 8; ++r) {
        int m = m0 + wave * 16 + r + lh * 8;  // psi row = f*128 + g
        int f = m >> 7, g = m & 127;
        psiT[base + (size_t)(g * d + v) * Kl + (f * d + u)] =
            (_Float16)(acc[r] * scale1);
      }
    }
  }
}

// ============================ Phase 2 ======================================
// One launch per l. grid(d /*N blocks*/, 4 /*M blocks of 64*/, d /*i*/),
// block 256 = 8 waves (2x4), wave tile 32x32 = 2x2 WMMA, K-chunks of 32,
// double-buffered LDS. B tiles use async global->LDS (no VGPR round trip).
#define P2_PITCH 40   // 32 + 8 halves padding

__global__ __launch_bounds__(256) void so3_kernel(
    const float*    __restrict__ x,     // [256,128,455]
    const _Float16* __restrict__ psiT,  // per-l W^T
    float*          __restrict__ out,   // [256,128,455]
    int d, int off, int base, unsigned magic, float scale)
{
  const int K   = 128 * d;
  const int nk  = 4 * d;                // K/32
  const int i_z = blockIdx.z;
  const int n0  = blockIdx.x * 128;
  const int m0  = blockIdx.y * 64;

  __shared__ _Float16 As[2][64 * P2_PITCH];   // [row b][k]
  __shared__ _Float16 Bs[2][128 * P2_PITCH];  // [col (g,v)][k]

  const int tid  = threadIdx.x;
  const int wave = tid >> 5;
  const int lane = tid & 31;
  const int lh   = lane >> 4;
  const int ll   = lane & 15;
  const int wm   = (wave >> 2) * 32;    // wave row offset in 64
  const int wn   = (wave & 3) * 32;     // wave col offset in 128

  const _Float16* gB = psiT + base;     // SGPR base for async loads

  auto loadA = [&](int s, int kc) {
    const int k0 = kc * 32;
    #pragma unroll
    for (int j = 0; j < 8; ++j) {
      int k   = tid & 31;
      int row = (tid >> 5) + 8 * j;           // 0..63
      int kk  = k0 + k;                       // (f,u) flat, < 1664
      int f   = (int)(((unsigned)kk * magic) >> 16);
      int u   = kk - f * d;
      int b   = m0 + row;
      float v = x[(size_t)((b << 7) + f) * IRREPS + off + u * d + i_z];
      As[s][row * P2_PITCH + k] = (_Float16)v;
    }
  };
  // 128 rows x 64B, 4 b128 chunks per row = 512 ops, 2 per thread.
  // ASYNCcnt-tracked; waited before the barrier that publishes the stage.
  auto loadB_async = [&](int s, int kc) {
    const int k0 = kc * 32;
    #pragma unroll
    for (int j = 0; j < 2; ++j) {
      int q = tid + 256 * j;                  // 0..511
      int n = q >> 2;                         // 0..127
      int c = q & 3;                          // 16B chunk in row
      unsigned voff = (unsigned)(((n0 + n) * K + k0) * 2 + c * 16);
      unsigned lptr = lds_off32(&Bs[s][n * P2_PITCH]) + (unsigned)(c * 16);
      asm volatile("global_load_async_to_lds_b128 %0, %1, %2"
                   :: "v"(lptr), "v"(voff), "s"(gB) : "memory");
    }
  };

  v8f acc[2][2] = {};
  loadA(0, 0);
  loadB_async(0, 0);
  asm volatile("s_wait_asynccnt 0x0" ::: "memory");
  __syncthreads();

  for (int kc = 0; kc < nk; ++kc) {
    const int s = kc & 1;
    if (kc + 1 < nk) { loadA(s ^ 1, kc + 1); loadB_async(s ^ 1, kc + 1); }

    v16h a[2], b[2];
    #pragma unroll
    for (int tm = 0; tm < 2; ++tm) {
      const _Float16* p = &As[s][(wm + tm * 16 + ll) * P2_PITCH + lh * 8];
      ((uint4*)&a[tm])[0] = *(const uint4*)(p);
      ((uint4*)&a[tm])[1] = *(const uint4*)(p + 16);
    }
    #pragma unroll
    for (int tn = 0; tn < 2; ++tn) {
      const _Float16* p = &Bs[s][(wn + tn * 16 + ll) * P2_PITCH + lh * 8];
      ((uint4*)&b[tn])[0] = *(const uint4*)(p);
      ((uint4*)&b[tn])[1] = *(const uint4*)(p + 16);
    }
    #pragma unroll
    for (int tm = 0; tm < 2; ++tm)
      #pragma unroll
      for (int tn = 0; tn < 2; ++tn)
        acc[tm][tn] = __builtin_amdgcn_wmma_f32_16x16x32_f16(
            false, a[tm], false, b[tn], (short)0, acc[tm][tn], false, false);

    asm volatile("s_wait_asynccnt 0x0" ::: "memory");
    __syncthreads();
  }

  // epilogue: out[b, g, off + v*d + i]
  #pragma unroll
  for (int tm = 0; tm < 2; ++tm) {
    #pragma unroll
    for (int tn = 0; tn < 2; ++tn) {
      int n = n0 + wn + tn * 16 + ll;         // (g,v) flat, < 1664
      int g = (int)(((unsigned)n * magic) >> 16);
      int v = n - g * d;
      #pragma unroll
      for (int r = 0; r < 8; ++r) {
        int b = m0 + wm + tm * 16 + r + lh * 8;
        out[(size_t)((b << 7) + g) * IRREPS + off + v * d + i_z] =
            acc[tm][tn][r] * scale;
      }
    }
  }
}

// ============================ launch =======================================
extern "C" void kernel_launch(void* const* d_in, const int* in_sizes, int n_in,
                              void* d_out, int out_size, void* d_ws, size_t ws_size,
                              hipStream_t stream) {
  const float* x  = (const float*)d_in[0];   // [256,128,455]
  const float* w  = (const float*)d_in[1];   // [128,128,512]
  const float* Dm = (const float*)d_in[2];   // [512,455]
  float* out      = (float*)d_out;
  _Float16* psiT  = (_Float16*)d_ws;         // needs 14.9 MB

  static const int OFF[8] = {0, 1, 10, 35, 84, 165, 286, 455};

  const size_t lds1 = (size_t)(128 + 16) * P1_PITCH * sizeof(_Float16);
  (void)hipFuncSetAttribute((const void*)psi_kernel,
                            hipFuncAttributeMaxDynamicSharedMemorySize,
                            (int)lds1);
  psi_kernel<<<128, 256, lds1, stream>>>(w, Dm, psiT);

  for (int l = 0; l <= 6; ++l) {
    const int d       = 2 * l + 1;
    const int off     = OFF[l];
    const int base    = OFF[l] * 16384;             // f16 elems into psiT
    const unsigned mg = (65536u + (unsigned)d - 1u) / (unsigned)d;
    const float scale = 1.0f / sqrtf(128.0f * (float)d);
    dim3 grid(d, 4, d);
    so3_kernel<<<grid, 256, 0, stream>>>(x, psiT, out, d, off, base, mg, scale);
  }
}
